// MultiHeadAttention_50929722196716
// MI455X (gfx1250) — matrix operand order repaired and checked by reading, not compiled, not run on hardware
//
#include <hip/hip_runtime.h>
#include <hip/hip_bf16.h>
#include <stdint.h>

#define H_DIM  1024
#define NHEAD  16
#define HD     64
#define BATCH  4
#define SEQ    2048
#define M_ROWS (BATCH * SEQ)

typedef __attribute__((ext_vector_type(16))) __bf16 v16bf;
typedef __attribute__((ext_vector_type(8)))  float  v8f;
typedef __attribute__((ext_vector_type(4)))  unsigned int v4u;
typedef __attribute__((ext_vector_type(8)))  int    v8i;
typedef __attribute__((ext_vector_type(4)))  int    v4i;

#if __has_builtin(__builtin_amdgcn_tensor_load_to_lds)
#define USE_TDM 1
#else
#define USE_TDM 0
#endif

static __device__ __forceinline__ v8f wmma_bf16(v16bf a, v16bf b, v8f c) {
    // D(f32 16x16) = A(bf16 16x32) * B(bf16 32x16) + C
    return __builtin_amdgcn_wmma_f32_16x16x32_bf16(
        /*neg_a=*/false, a, /*neg_b=*/false, b,
        /*c_mod=*/(short)0, c, /*reuse_a=*/false, /*reuse_b=*/false);
}

#if USE_TDM
// TDM 2D tile load: rows x 64 bf16 elements, global row stride H_DIM elems,
// LDS destination padded to 160B row stride via pad_interval/pad_amount.
static __device__ __forceinline__ void tdm_load_2d(const void* gptr,
                                                   unsigned int lds_off,
                                                   int rows) {
    unsigned long long ga = (unsigned long long)(uintptr_t)gptr;
    v4u g0;
    g0.x = 1u;                                    // count=1, user descriptor
    g0.y = lds_off;                               // LDS byte address
    g0.z = (unsigned int)ga;                      // global_addr[31:0]
    g0.w = (unsigned int)((ga >> 32) & 0x1FFFFFFu) | 0x80000000u; // [56:32] | type=2
    v8i g1;
    // data_size=1 (2B) | pad_enable | pad_interval=4 (32 DWORDs = 128B row)
    // | pad_amount=7 (8 DWORDs = 32B) -> LDS row stride 160B
    g1[0] = (1 << 16) | (1 << 20) | (4 << 22) | (7 << 25);
    g1[1] = 0;                                    // barrier addr 0; tensor_dim0 lo16 (of 1M) = 0
    g1[2] = 16;                                   // tensor_dim0 hi16 = 16 (1M); dim1 lo16 = 0
    g1[3] = 16 | (64 << 16);                      // tensor_dim1 hi16 = 16 (1M); tile_dim0 = 64
    g1[4] = rows;                                 // tile_dim1 = rows; tile_dim2 = 0
    g1[5] = H_DIM;                                // tensor_dim0_stride = 1024 elems
    g1[6] = 0;
    g1[7] = 0;
    v4i z4 = {0, 0, 0, 0};
    v8i z8 = {0, 0, 0, 0, 0, 0, 0, 0};
    __builtin_amdgcn_tensor_load_to_lds(g0, g1, z4, z4, z8, 0);
}
#endif

// ---------------------------------------------------------------------------
// Tiled GEMM: C[M,N] = A[M,K] @ W[N,K]^T + bias[N]
// Block tile 128x128, 256 threads = 8 waves laid out 4(m) x 2(n),
// each wave computes 32x64 via 2x4 16x16 f32 accumulators, K-step 32.
// A is f32 or bf16 (template), W is f32, both converted to bf16 in LDS.
// ---------------------------------------------------------------------------
template <bool A_F32, bool OUT_F32>
__global__ __launch_bounds__(256) void gemm_wmma(
    const void* __restrict__ Ap, const float* __restrict__ W,
    const float* __restrict__ bias, void* __restrict__ Cp,
    int M, int N, int K)
{
    __shared__ __bf16 As[128][40];   // stride 80B (16B aligned)
    __shared__ __bf16 Ws[128][40];

    const int tid   = threadIdx.x;
    const int lane  = tid & 31;
    const int wave  = tid >> 5;
    const int wm    = wave & 3;      // 0..3
    const int wn    = wave >> 2;     // 0..1
    const int lrow  = lane & 15;
    const int lhalf = lane >> 4;
    const int m0    = blockIdx.y * 128;
    const int n0    = blockIdx.x * 128;

    v8f acc[2][4] = {};

    for (int k0 = 0; k0 < K; k0 += 32) {
        __syncthreads();
        // --- stage A tile (128x32) as bf16 ---
        if (A_F32) {
            const float* Af = (const float*)Ap;
            #pragma unroll
            for (int ii = 0; ii < 4; ++ii) {
                int t = tid + ii * 256;
                int r = t >> 3, c = (t & 7) << 2;
                float4 v = *(const float4*)(Af + (size_t)(m0 + r) * K + k0 + c);
                As[r][c + 0] = (__bf16)v.x; As[r][c + 1] = (__bf16)v.y;
                As[r][c + 2] = (__bf16)v.z; As[r][c + 3] = (__bf16)v.w;
            }
            if (k0 + 32 < K)   // prefetch next A k-tile into L2
                __builtin_prefetch(Af + (size_t)(m0 + (tid >> 3)) * K + k0 + 32, 0, 1);
        } else {
            const __bf16* Ab = (const __bf16*)Ap;
            #pragma unroll
            for (int ii = 0; ii < 2; ++ii) {
                int t = tid + ii * 256;
                int r = t >> 2, c = (t & 3) << 3;
                *(uint4*)&As[r][c] = *(const uint4*)(Ab + (size_t)(m0 + r) * K + k0 + c);
            }
            if (k0 + 32 < K)
                __builtin_prefetch((const __bf16*)Ap + (size_t)(m0 + (tid >> 2)) * K + k0 + 32, 0, 1);
        }
        // --- stage W tile (rows n0..n0+127, cols k0..k0+31) as bf16 ---
        #pragma unroll
        for (int ii = 0; ii < 4; ++ii) {
            int t = tid + ii * 256;
            int r = t >> 3, c = (t & 7) << 2;
            float4 v = *(const float4*)(W + (size_t)(n0 + r) * K + k0 + c);
            Ws[r][c + 0] = (__bf16)v.x; Ws[r][c + 1] = (__bf16)v.y;
            Ws[r][c + 2] = (__bf16)v.z; Ws[r][c + 3] = (__bf16)v.w;
        }
        if (k0 + 32 < K)       // prefetch next W k-tile into L2
            __builtin_prefetch(W + (size_t)(n0 + (tid >> 3)) * K + k0 + 32, 0, 1);
        __syncthreads();

        // --- A fragments: 16x32, lane row = lrow, K half per lane-group ---
        v16bf afr[2], bfr[4];
        #pragma unroll
        for (int i = 0; i < 2; ++i) {
            int row = wm * 32 + i * 16 + lrow;
            int kb  = lhalf * 8;
            #pragma unroll
            for (int e = 0; e < 16; ++e)
                afr[i][e] = As[row][e + (e & 8) + kb];
        }
        // --- B fragments: 32x16, B[k][n] = W[n][k]; lane col = lrow ---
        #pragma unroll
        for (int j = 0; j < 4; ++j) {
            int nn = wn * 64 + j * 16 + lrow;
            int kb = lhalf * 8;
            #pragma unroll
            for (int e = 0; e < 16; ++e)
                bfr[j][e] = Ws[nn][kb + e + (e & 8)];
        }
        #pragma unroll
        for (int i = 0; i < 2; ++i)
            #pragma unroll
            for (int j = 0; j < 4; ++j)
                acc[i][j] = wmma_bf16(afr[i], bfr[j], acc[i][j]);
    }

    // --- epilogue: C layout = lane col lrow, rows r + 8*lhalf ---
    #pragma unroll
    for (int i = 0; i < 2; ++i) {
        #pragma unroll
        for (int j = 0; j < 4; ++j) {
            int col  = n0 + wn * 64 + j * 16 + lrow;
            float bv = bias[col];
            #pragma unroll
            for (int r = 0; r < 8; ++r) {
                int row = m0 + wm * 32 + i * 16 + r + lhalf * 8;
                float v = acc[i][j][r] + bv;
                if (OUT_F32) ((float*)Cp)[(size_t)row * N + col] = v;
                else         ((__bf16*)Cp)[(size_t)row * N + col] = (__bf16)v;
            }
        }
    }
}

// ---------------------------------------------------------------------------
// Flash attention: one workgroup (128 thr / 4 waves) per (b, h, 64 q-rows).
// Q/K/V tiles are DMA'd into LDS by the Tensor Data Mover (wave 0 issues,
// TENSORcnt wait + barrier publishes); scores + PV via bf16 WMMA, online
// softmax with wave32 lane reductions.
// ---------------------------------------------------------------------------
__global__ __launch_bounds__(128) void flash_attn(
    const __bf16* __restrict__ Q, const __bf16* __restrict__ Kp,
    const __bf16* __restrict__ V, const float* __restrict__ amask,
    __bf16* __restrict__ ctx)
{
    __shared__ __bf16 Qs[64][80];    // 160B row stride (TDM pad target)
    __shared__ __bf16 Ks[32][80];
    __shared__ __bf16 Vs[32][80];
    __shared__ __bf16 Ps[64][40];    // P repack staging (wave-private rows)
    __shared__ float  Ms[32];

    const int tid   = threadIdx.x;
    const int lane  = tid & 31;
    const int wave  = tid >> 5;
    const int lrow  = lane & 15;
    const int lhalf = lane >> 4;
    const int qblk  = blockIdx.x;
    const int h     = blockIdx.y;
    const int b     = blockIdx.z;
    const size_t headoff = (size_t)h * HD;
    const float scale = 0.125f;      // 1/sqrt(64)

    // --- load 64x64 Q block into LDS ---
    const __bf16* qbase = Q + ((size_t)b * SEQ + (size_t)qblk * 64) * H_DIM + headoff;
#if USE_TDM
    if (wave == 0) {
        tdm_load_2d(qbase, (unsigned int)(uintptr_t)&Qs[0][0], 64);
        __builtin_amdgcn_s_wait_tensorcnt(0);
    }
#else
    #pragma unroll
    for (int ii = 0; ii < 4; ++ii) {
        int t = tid + ii * 128;
        int r = t >> 3, c = (t & 7) << 3;
        *(uint4*)&Qs[r][c] = *(const uint4*)(qbase + (size_t)r * H_DIM + c);
    }
#endif
    __syncthreads();

    // --- resident Q A-fragments (two HD-halves of 32) ---
    v16bf qfr[2];
    #pragma unroll
    for (int hh = 0; hh < 2; ++hh) {
        int row = wave * 16 + lrow;
        int kb  = lhalf * 8;
        #pragma unroll
        for (int e = 0; e < 16; ++e)
            qfr[hh][e] = Qs[row][hh * 32 + e + (e & 8) + kb];
    }

    float m_i[8], l_i[8];
    #pragma unroll
    for (int r = 0; r < 8; ++r) { m_i[r] = -3.0e38f; l_i[r] = 0.0f; }
    v8f accO[4] = {};

    for (int kb0 = 0; kb0 < SEQ / 32; ++kb0) {
        __syncthreads();
        const __bf16* kbase = Kp + ((size_t)b * SEQ + (size_t)kb0 * 32) * H_DIM + headoff;
        const __bf16* vbase = V  + ((size_t)b * SEQ + (size_t)kb0 * 32) * H_DIM + headoff;
#if USE_TDM
        if (wave == 0) {
            tdm_load_2d(kbase, (unsigned int)(uintptr_t)&Ks[0][0], 32);
            tdm_load_2d(vbase, (unsigned int)(uintptr_t)&Vs[0][0], 32);
            __builtin_amdgcn_s_wait_tensorcnt(0);
        }
#else
        #pragma unroll
        for (int ii = 0; ii < 2; ++ii) {
            int t = tid + ii * 128;
            int r = t >> 3, c = (t & 7) << 3;
            *(uint4*)&Ks[r][c] = *(const uint4*)(kbase + (size_t)r * H_DIM + c);
            *(uint4*)&Vs[r][c] = *(const uint4*)(vbase + (size_t)r * H_DIM + c);
        }
#endif
        if (tid < 32) Ms[tid] = amask[(size_t)b * SEQ + (size_t)kb0 * 32 + tid];
        __syncthreads();

        // --- scores: 16 q-rows x 32 keys as two 16x16 tiles ---
        v8f sc[2];
        #pragma unroll
        for (int ns = 0; ns < 2; ++ns) {
            v16bf bk0, bk1;   // B[k=hd][n=key]: key = ns*16 + lrow
            #pragma unroll
            for (int e = 0; e < 16; ++e) {
                bk0[e] = Ks[ns * 16 + lrow][lhalf * 8 + e + (e & 8)];
                bk1[e] = Ks[ns * 16 + lrow][32 + lhalf * 8 + e + (e & 8)];
            }
            v8f c0 = {};
            c0 = wmma_bf16(qfr[0], bk0, c0);
            c0 = wmma_bf16(qfr[1], bk1, c0);
            sc[ns] = c0;
        }

        const float mc0 = (1.0f - Ms[lrow])      * -10000.0f;
        const float mc1 = (1.0f - Ms[16 + lrow]) * -10000.0f;

        // --- online softmax over 32-key block (row reduce across 16 lanes) ---
        float alpha[8];
        #pragma unroll
        for (int r = 0; r < 8; ++r) {
            float s0 = sc[0][r] * scale + mc0;
            float s1 = sc[1][r] * scale + mc1;
            float rm = fmaxf(s0, s1);
            #pragma unroll
            for (int off = 1; off < 16; off <<= 1)
                rm = fmaxf(rm, __shfl_xor(rm, off, 32));
            float mnew = fmaxf(m_i[r], rm);
            float a  = __expf(m_i[r] - mnew);
            float p0 = __expf(s0 - mnew);
            float p1 = __expf(s1 - mnew);
            float rs = p0 + p1;
            #pragma unroll
            for (int off = 1; off < 16; off <<= 1)
                rs += __shfl_xor(rs, off, 32);
            l_i[r] = l_i[r] * a + rs;
            m_i[r] = mnew;
            alpha[r] = a;
            int prow = wave * 16 + r + lhalf * 8;   // wave-private rows
            Ps[prow][lrow]      = (__bf16)p0;
            Ps[prow][16 + lrow] = (__bf16)p1;
        }
        #pragma unroll
        for (int hs = 0; hs < 4; ++hs)
            #pragma unroll
            for (int r = 0; r < 8; ++r)
                accO[hs][r] *= alpha[r];

        // --- repack P (C layout -> A layout) via LDS, then PV WMMAs ---
        v16bf pfr;
        {
            int kb = lhalf * 8;
            #pragma unroll
            for (int e = 0; e < 16; ++e)
                pfr[e] = Ps[wave * 16 + lrow][e + (e & 8) + kb];
        }
        #pragma unroll
        for (int hs = 0; hs < 4; ++hs) {
            v16bf vf;   // B[k=key][n=hd col]
            #pragma unroll
            for (int e = 0; e < 16; ++e)
                vf[e] = Vs[lhalf * 8 + e + (e & 8)][hs * 16 + lrow];
            accO[hs] = wmma_bf16(pfr, vf, accO[hs]);
        }
    }

    // --- normalize and write ctx (bf16, [B,S,H] with head offset) ---
    #pragma unroll
    for (int r = 0; r < 8; ++r) {
        float inv = 1.0f / l_i[r];
        int qrow  = qblk * 64 + wave * 16 + r + lhalf * 8;
        size_t base = ((size_t)b * SEQ + qrow) * H_DIM + headoff;
        #pragma unroll
        for (int hs = 0; hs < 4; ++hs)
            ctx[base + hs * 16 + lrow] = (__bf16)(accO[hs][r] * inv);
    }
}

// ---------------------------------------------------------------------------
extern "C" void kernel_launch(void* const* d_in, const int* in_sizes, int n_in,
                              void* d_out, int out_size, void* d_ws, size_t ws_size,
                              hipStream_t stream) {
    const float* query = (const float*)d_in[0];
    const float* key_  = (const float*)d_in[1];
    const float* value = (const float*)d_in[2];
    const float* amask = (const float*)d_in[3];
    const float* Wq = (const float*)d_in[4];
    const float* bq = (const float*)d_in[5];
    const float* Wk = (const float*)d_in[6];
    const float* bk = (const float*)d_in[7];
    const float* Wv = (const float*)d_in[8];
    const float* bv = (const float*)d_in[9];
    const float* Wo = (const float*)d_in[10];
    const float* bo = (const float*)d_in[11];

    // workspace: Q, K, V, ctx as bf16 [B*S, H] = 16 MiB each (64 MiB total)
    const size_t elems = (size_t)M_ROWS * H_DIM;
    char* wsb = (char*)d_ws;
    __bf16* Qb  = (__bf16*)(wsb + 0 * elems * 2);
    __bf16* Kb  = (__bf16*)(wsb + 1 * elems * 2);
    __bf16* Vb  = (__bf16*)(wsb + 2 * elems * 2);
    __bf16* Ctx = (__bf16*)(wsb + 3 * elems * 2);

    dim3 gblk(256);
    dim3 ggrid(H_DIM / 128, M_ROWS / 128);   // (8, 64)
    gemm_wmma<true,  false><<<ggrid, gblk, 0, stream>>>(query, Wq, bq, Qb,  M_ROWS, H_DIM, H_DIM);
    gemm_wmma<true,  false><<<ggrid, gblk, 0, stream>>>(key_,  Wk, bk, Kb,  M_ROWS, H_DIM, H_DIM);
    gemm_wmma<true,  false><<<ggrid, gblk, 0, stream>>>(value, Wv, bv, Vb,  M_ROWS, H_DIM, H_DIM);

    dim3 agrid(SEQ / 64, NHEAD, BATCH);      // (32, 16, 4)
    flash_attn<<<agrid, dim3(128), 0, stream>>>(Qb, Kb, Vb, amask, Ctx);

    gemm_wmma<false, true><<<ggrid, gblk, 0, stream>>>(Ctx, Wo, bo, (float*)d_out,
                                                       M_ROWS, H_DIM, H_DIM);
}
